// StyleEncoder_29420525977830
// MI455X (gfx1250) — compile-verified, matrix operand order checked by reading, not run on hardware
//
#include <hip/hip_runtime.h>
#include <hip/hip_bf16.h>

// LSTM: B=64, T=2048, IN=200, H=100. Output h_T [64,100] f32.
// Phase 1: xg[t][b][g] = x@W_ih^T + (b_ih+b_hh)  -> d_ws (needs 2048*64*400*4 = 210MB)
// Phase 2: persistent single-workgroup recurrence, bf16 WMMA + async global->LDS
//          double buffering of the xg[t] slab (ASYNCcnt path).

#define T_STEPS 2048
#define BATCH   64
#define IN_SZ   200
#define HID     100
#define G4      400           // 4*HID
#define SLAB    (BATCH * G4)  // 25600 f32 = 102400 B per timestep

typedef __attribute__((ext_vector_type(16))) __bf16 v16bf;
typedef __attribute__((ext_vector_type(8)))  float  v8f;

union Frag {
    v16bf  v;
    __bf16 e[16];
    uint4  u4[2];
};

__device__ __forceinline__ float sigmoidf_(float x) {
    return 1.0f / (1.0f + __expf(-x));
}

__device__ __forceinline__ void cvt8_global(const float* __restrict__ p, __bf16* dst) {
    float4 a = *(const float4*)(p);
    float4 b = *(const float4*)(p + 4);
    dst[0] = (__bf16)a.x; dst[1] = (__bf16)a.y; dst[2] = (__bf16)a.z; dst[3] = (__bf16)a.w;
    dst[4] = (__bf16)b.x; dst[5] = (__bf16)b.y; dst[6] = (__bf16)b.z; dst[7] = (__bf16)b.w;
}

__device__ __forceinline__ void zero8(__bf16* dst) {
    #pragma unroll
    for (int i = 0; i < 8; ++i) dst[i] = (__bf16)0.0f;
}

// Async copy 128B/thread of a timestep slab into LDS (8 x b128, ASYNCcnt += 8).
// INST_OFFSET is added to BOTH the LDS and global addresses per the ISA pseudocode.
#define ASYNC_LD(OFF)                                                        \
    asm volatile("global_load_async_to_lds_b128 %0, %1, off offset:" #OFF    \
                 :: "v"(ldsOff), "v"(gsrc) : "memory")

__device__ __forceinline__ void async_slab_copy(const float* gsrc, unsigned ldsOff) {
    ASYNC_LD(0);  ASYNC_LD(16); ASYNC_LD(32); ASYNC_LD(48);
    ASYNC_LD(64); ASYNC_LD(80); ASYNC_LD(96); ASYNC_LD(112);
}

__device__ __forceinline__ void wait_asynccnt_le8() {
#if __has_builtin(__builtin_amdgcn_s_wait_asynccnt)
    __builtin_amdgcn_s_wait_asynccnt(8);
#else
    asm volatile("s_wait_asynccnt 0x8" ::: "memory");
#endif
}
__device__ __forceinline__ void wait_asynccnt_0() {
#if __has_builtin(__builtin_amdgcn_s_wait_asynccnt)
    __builtin_amdgcn_s_wait_asynccnt(0);
#else
    asm volatile("s_wait_asynccnt 0x0" ::: "memory");
#endif
}

// ---------------- Phase 1: input projection GEMM -----------------------------
// M = T*B = 131072 rows (r = t*64 + b), N = 400, K = 200 (7 chunks of 32, padded).
__global__ __launch_bounds__(256) void lstm_xproj_wmma(
    const float* __restrict__ x, const float* __restrict__ W_ih,
    const float* __restrict__ b_ih, const float* __restrict__ b_hh,
    float* __restrict__ xg)
{
    const int lane = threadIdx.x & 31;
    const int l15  = lane & 15;
    const int hi8  = (lane >> 4) * 8;

    const int gw    = blockIdx.x * 8 + (threadIdx.x >> 5);
    const int ntile = gw % 25;
    const int mtile = gw / 25;
    const int nbase = ntile * 16;
    const int mbase = mtile * 16;

    const int r  = mbase + l15;
    const int tt = r >> 6;
    const int bb = r & 63;
    const float* xrow = x + ((size_t)bb * T_STEPS + tt) * IN_SZ;
    const int n = nbase + l15;
    const float* wrow = W_ih + (size_t)n * IN_SZ;
    const float bias  = b_ih[n] + b_hh[n];

    v8f c = {};
    #pragma unroll
    for (int kc = 0; kc < 7; ++kc) {
        Frag af, bf;
        const int k0 = kc * 32 + hi8;
        const int k1 = k0 + 16;
        if (k0 + 8 <= IN_SZ) cvt8_global(xrow + k0, &af.e[0]); else zero8(&af.e[0]);
        if (k1 + 8 <= IN_SZ) cvt8_global(xrow + k1, &af.e[8]); else zero8(&af.e[8]);
        if (k0 + 8 <= IN_SZ) cvt8_global(wrow + k0, &bf.e[0]); else zero8(&bf.e[0]);
        if (k1 + 8 <= IN_SZ) cvt8_global(wrow + k1, &bf.e[8]); else zero8(&bf.e[8]);
        c = __builtin_amdgcn_wmma_f32_16x16x32_bf16(false, af.v, false, bf.v,
                                                    (short)0, c, false, false);
    }

    #pragma unroll
    for (int v = 0; v < 8; ++v) {
        const int row = mbase + v + hi8;
        xg[(size_t)row * G4 + nbase + l15] = c[v] + bias;
    }
}

// ---------------- Phase 2: persistent recurrence -----------------------------
// 800 threads = 25 waves; wave w owns N-tile [16w,16w+16).
// xg[t] slabs are double-buffered in LDS via async global->LDS DMA; gate
// pre-activations overwrite the current slab in place (same lane reads then
// writes each (b,n) location). W_hh lives in registers as bf16 B-fragments.
__global__ __launch_bounds__(800) void lstm_recur_wmma(
    const float* __restrict__ W_hh, const float* __restrict__ xg,
    float* __restrict__ out)
{
    __shared__ __bf16 hLds[BATCH][128];     // 16 KB, K padded with zeros
    __shared__ float  xgBuf[2][SLAB];       // 2 x 100 KB staging + gates

    const int tid   = threadIdx.x;
    const int lane  = tid & 31;
    const int l15   = lane & 15;
    const int hi8   = (lane >> 4) * 8;
    const int w     = tid >> 5;
    const int nbase = w * 16;
    const int n     = nbase + l15;

    for (int i = tid; i < BATCH * 128; i += 800)
        ((__bf16*)hLds)[i] = (__bf16)0.0f;

    // W_hh B-fragments: element e of lane -> K = kc*32 + hi8 + (e<8 ? e : e+8).
    Frag bfr[4];
    const float* wrow = W_hh + (size_t)n * HID;
    #pragma unroll
    for (int kc = 0; kc < 4; ++kc) {
        #pragma unroll
        for (int e = 0; e < 16; ++e) {
            const int k = kc * 32 + hi8 + ((e < 8) ? e : (e + 8));
            bfr[kc].e[e] = (k < HID) ? (__bf16)wrow[k] : (__bf16)0.0f;
        }
    }

    const int base = tid * 8;
    float creg[8], hreg[8];
    int bArr[8], jArr[8];
    #pragma unroll
    for (int k = 0; k < 8; ++k) {
        creg[k] = 0.0f; hreg[k] = 0.0f;
        bArr[k] = (base + k) / HID;
        jArr[k] = (base + k) % HID;
    }

    // Kick off slab 0 (each thread owns a disjoint 128B stripe).
    async_slab_copy(xg + (size_t)tid * 32,
                    (unsigned)(size_t)(&xgBuf[0][0]) + (unsigned)tid * 128u);

    for (int t = 0; t < T_STEPS; ++t) {
        const int cur = t & 1;

        if (t + 1 < T_STEPS) {
            // Stream next slab while this step computes; ASYNCcnt in-order
            // completion means <=8 outstanding guarantees slab(t) has landed.
            async_slab_copy(xg + (size_t)(t + 1) * SLAB + (size_t)tid * 32,
                            (unsigned)(size_t)(&xgBuf[cur ^ 1][0]) + (unsigned)tid * 128u);
            wait_asynccnt_le8();
        } else {
            wait_asynccnt_0();
        }
        __syncthreads();   // slab(t) visible to all waves

        float* slab = &xgBuf[cur][0];

        // gates = xg[t] + h @ W_hh^T  (C initialized from the LDS slab)
        #pragma unroll
        for (int m = 0; m < 4; ++m) {
            const int mbase = m * 16;
            v8f c;
            #pragma unroll
            for (int v = 0; v < 8; ++v)
                c[v] = slab[(mbase + v + hi8) * G4 + n];
            #pragma unroll
            for (int kc = 0; kc < 4; ++kc) {
                Frag af;
                const __bf16* hrow = &hLds[mbase + l15][0];
                af.u4[0] = *(const uint4*)(hrow + kc * 32 + hi8);
                af.u4[1] = *(const uint4*)(hrow + kc * 32 + hi8 + 16);
                c = __builtin_amdgcn_wmma_f32_16x16x32_bf16(false, af.v, false, bfr[kc].v,
                                                            (short)0, c, false, false);
            }
            #pragma unroll
            for (int v = 0; v < 8; ++v)
                slab[(mbase + v + hi8) * G4 + n] = c[v];   // overwrite in place
        }
        __syncthreads();

        // pointwise LSTM cell update (gate order i,f,g,o)
        #pragma unroll
        for (int k = 0; k < 8; ++k) {
            const int b = bArr[k], j = jArr[k];
            const float ig = sigmoidf_(slab[b * G4 + j]);
            const float fg = sigmoidf_(slab[b * G4 + HID + j]);
            const float gg = tanhf(slab[b * G4 + 2 * HID + j]);
            const float og = sigmoidf_(slab[b * G4 + 3 * HID + j]);
            const float cc = fg * creg[k] + ig * gg;
            creg[k] = cc;
            const float hh = og * tanhf(cc);
            hreg[k] = hh;
            hLds[b][j] = (__bf16)hh;
        }
        __syncthreads();   // h ready; safe to let next step's DMA reuse buf
    }

    #pragma unroll
    for (int k = 0; k < 8; ++k)
        out[bArr[k] * HID + jArr[k]] = hreg[k];
}

extern "C" void kernel_launch(void* const* d_in, const int* in_sizes, int n_in,
                              void* d_out, int out_size, void* d_ws, size_t ws_size,
                              hipStream_t stream) {
    (void)in_sizes; (void)n_in; (void)out_size; (void)ws_size;
    const float* x    = (const float*)d_in[0];
    const float* W_ih = (const float*)d_in[1];
    const float* W_hh = (const float*)d_in[2];
    const float* b_ih = (const float*)d_in[3];
    const float* b_hh = (const float*)d_in[4];
    float* out = (float*)d_out;
    float* xg  = (float*)d_ws;   // 2048*64*400 f32 = 209,715,200 bytes

    lstm_xproj_wmma<<<dim3(25600), dim3(256), 0, stream>>>(x, W_ih, b_ih, b_hh, xg);
    lstm_recur_wmma<<<dim3(1), dim3(800), 0, stream>>>(W_hh, xg, out);
}
